// VMamba_v4_84490596647641
// MI455X (gfx1250) — compile-verified, matrix operand order checked by reading, not run on hardware
//
#include <hip/hip_runtime.h>
#include <math.h>

// ---------------- constants (match reference) ----------------
#define B0 2
#define SS 32
#define LL (SS*SS + 2)       // 1026
#define DM 256
#define ED 512
#define NN 16
#define RR 16
#define KK 4
#define MROWS (B0*LL)        // 2052

typedef __attribute__((ext_vector_type(16))) _Float16 v16h;
typedef __attribute__((ext_vector_type(8)))  _Float16 v8h;
typedef __attribute__((ext_vector_type(8)))  float    v8f;
typedef __attribute__((ext_vector_type(4)))  unsigned int v4u;
typedef __attribute__((ext_vector_type(8)))  int      v8i;
typedef __attribute__((ext_vector_type(4)))  int      v4i;

// ---------------- permutations (all involutions) ----------------
__device__ __forceinline__ int perm_idx(int s, int p) {
    if (s & 1) p = LL - 1 - p;          // branches b, d reverse first
    if (s >= 2) {                        // branches c, d apply cross
        if (p > 0 && p < LL - 1) {
            int m = p - 1;
            int r = m >> 5;
            int c = m & 31;
            p = 1 + (c << 5) + r;
        }
    }
    return p;
}

// ---------------- small utility kernels ----------------
__global__ void cvt_f32_f16(const float* __restrict__ src,
                            _Float16* __restrict__ dst, int n) {
    int i = blockIdx.x * blockDim.x + threadIdx.x;
    if (i < n) dst[i] = (_Float16)src[i];
}

__global__ void pad_f32_f16(const float* __restrict__ src,
                            _Float16* __restrict__ dst,
                            int rows, int src_ld, int copy, int dst_ld) {
    int i = blockIdx.x * blockDim.x + threadIdx.x;
    if (i >= rows * dst_ld) return;
    int r = i / dst_ld;
    int c = i - r * dst_ld;
    dst[i] = (c < copy) ? (_Float16)src[r * src_ld + c] : (_Float16)0.0f;
}

// yh = f16( 0.25 * (g0 + g1 + g2 + g3) ), branch stride n
__global__ void combine_cvt(const float* __restrict__ g,
                            _Float16* __restrict__ yh, int n) {
    int i = blockIdx.x * blockDim.x + threadIdx.x;
    if (i < n) {
        float v = 0.25f * (g[i] + g[i + n] + g[i + 2 * (size_t)n] + g[i + 3 * (size_t)n]);
        yh[i] = (_Float16)v;
    }
}

// ---------------- fragment helpers ----------------
__device__ __forceinline__ v16h make_frag(v8h lo, v8h hi) {
    v16h v;
#pragma unroll
    for (int j = 0; j < 8; ++j) { v[j] = lo[j]; v[j + 8] = hi[j]; }
    return v;
}

__device__ __forceinline__ void store_tile(float* __restrict__ C, v8f acc,
                                           int M, int N, int mtile, int col,
                                           const float* __restrict__ bias, int act,
                                           int lane) {
    const int mbase = mtile * 16 + (lane >> 4) * 8;
    float b = (act && bias) ? bias[col] : 0.0f;
#pragma unroll
    for (int v = 0; v < 8; ++v) {
        int mm = mbase + v;
        if (mm < M) {
            float val = acc[v];
            if (act) {                        // softplus(val + dt_b)
                val += b;
                val = (val > 20.0f) ? val : log1pf(__expf(val));
            }
            C[(size_t)mm * N + col] = val;
        }
    }
}

// ---------------- simple WMMA GEMM (16x16 tile / wave), for N % 64 != 0 ----
__global__ __launch_bounds__(32)
void gemm_wmma_f16(const _Float16* __restrict__ A,
                   const _Float16* __restrict__ W,
                   float* __restrict__ C,
                   int M, int N, int K,
                   const float* __restrict__ bias, int act) {
    const int lane = threadIdx.x & 31;
    const int r    = lane & 15;
    const int kh   = lane >> 4;
    const int ntile = blockIdx.x;
    const int mtile = blockIdx.y;

    int m = mtile * 16 + r; if (m >= M) m = M - 1;   // clamp: EXEC stays full
    const int n = ntile * 16 + r;

    const _Float16* pa = A + (size_t)m * K + kh * 8;
    const _Float16* pb = W + (size_t)n * K + kh * 8;

    v8f acc = {};
    for (int k0 = 0; k0 < K; k0 += 32) {
        v8h alo = *(const v8h*)(pa + k0);
        v8h ahi = *(const v8h*)(pa + k0 + 16);
        v8h blo = *(const v8h*)(pb + k0);
        v8h bhi = *(const v8h*)(pb + k0 + 16);
        if (k0 + 32 < K) {
            __builtin_prefetch(pa + k0 + 32, 0, 1);
            __builtin_prefetch(pb + k0 + 32, 0, 1);
        }
        acc = __builtin_amdgcn_wmma_f32_16x16x32_f16(
            false, make_frag(alo, ahi), false, make_frag(blo, bhi),
            (short)0, acc, false, false);
    }
    store_tile(C, acc, M, N, mtile, ntile * 16 + (lane & 15), bias, act, lane);
}

// ---------------- TDM-staged WMMA GEMM ----------------
// Block = 4 waves, computes 16 x 256 of C; each wave does a 16x64 tile.
// The shared 16x32 A tile is DMA'd global->LDS by the Tensor Data Mover
// (double buffered), consumed via ds_load_b128; B comes straight from L2.
__global__ __launch_bounds__(128)
void gemm_wmma_tdm(const _Float16* __restrict__ A,
                   const _Float16* __restrict__ W,
                   float* __restrict__ C,
                   int M, int N, int K,
                   const float* __restrict__ bias, int act) {
    __shared__ _Float16 smA[2][16 * 32];   // 2 x 1KB double buffer

    const int tid  = threadIdx.x;
    const int lane = tid & 31;
    const int wave = tid >> 5;
    const int r    = lane & 15;
    const int kh   = lane >> 4;
    const int mtile = blockIdx.y;
    const int colbase = blockIdx.x * 256 + wave * 64;
    const int niter = K >> 5;

    const _Float16* pb[4];
#pragma unroll
    for (int t = 0; t < 4; ++t)
        pb[t] = W + (size_t)(colbase + t * 16 + r) * K + kh * 8;

    v8f acc[4] = {v8f{}, v8f{}, v8f{}, v8f{}};

    // ----- Tensor DMA descriptor (2-D tile: 32 halfs x 16 rows) -----
    const unsigned long long gbase =
        (unsigned long long)(uintptr_t)(A + (size_t)(mtile * 16) * K);
    v4u g0;
    v8i g1;
    v4i g2 = {0, 0, 0, 0}, g3 = {0, 0, 0, 0};
    v8i g4 = {0, 0, 0, 0, 0, 0, 0, 0};       // trailing group (unused, zero)
    g1[0] = (int)(1u << 16);                 // data_size = 1 -> 2 bytes
    g1[1] = 0;                               // atomic_barrier=0, tensor_dim0 lo16 (dim0 = 1<<30)
    g1[2] = (int)0x00004000u;                // tensor_dim0 hi16, tensor_dim1 lo16
    g1[3] = (int)(0x4000u | (32u << 16));    // tensor_dim1 hi16, tile_dim0 = 32
    g1[4] = (int)16u;                        // tile_dim1 = 16, tile_dim2 = 0
    g1[5] = (int)K;                          // tensor_dim0_stride = K elements
    g1[6] = 0;
    g1[7] = 0;

    auto issue = [&](int it, int buf) {
        unsigned long long ga = gbase + (unsigned long long)it * 64ull; // 32 halfs
        g0[0] = 1u;                                            // count = 1
        g0[1] = (unsigned)(uintptr_t)&smA[buf][0];             // LDS byte addr (addr[31:0])
        g0[2] = (unsigned)(ga & 0xFFFFFFFFull);
        g0[3] = (unsigned)((ga >> 32) & 0x01FFFFFFull) | (2u << 30);  // type=2
        __builtin_amdgcn_tensor_load_to_lds(g0, g1, g2, g3, g4, 0);
    };

    if (tid < 32) issue(0, 0);               // wave0 drives the TDM

    for (int i = 0; i < niter; ++i) {
        const int buf = i & 1;
        if (tid < 32) __builtin_amdgcn_s_wait_tensorcnt(0);
        __syncthreads();                     // tile i resident for all waves
        if (tid < 32 && (i + 1) < niter) issue(i + 1, buf ^ 1);

        // A fragment from LDS (ds_load_b128 x2)
        v8h alo = *(const v8h*)&smA[buf][r * 32 + kh * 8];
        v8h ahi = *(const v8h*)&smA[buf][r * 32 + 16 + kh * 8];
        v16h av = make_frag(alo, ahi);

        const int k0 = i * 32;
#pragma unroll
        for (int t = 0; t < 4; ++t) {
            v8h blo = *(const v8h*)(pb[t] + k0);
            v8h bhi = *(const v8h*)(pb[t] + k0 + 16);
            acc[t] = __builtin_amdgcn_wmma_f32_16x16x32_f16(
                false, av, false, make_frag(blo, bhi),
                (short)0, acc[t], false, false);
        }
        __syncthreads();                     // buf fully consumed before rewrite
    }

#pragma unroll
    for (int t = 0; t < 4; ++t)
        store_tile(C, acc[t], M, N, mtile, colbase + t * 16 + (lane & 15),
                   bias, act, lane);
}

// ---------------- depthwise causal conv (K=4) + SiLU ----------------
__global__ void conv_silu_kernel(const float* __restrict__ xz,
                                 const float* __restrict__ cw,
                                 const float* __restrict__ cb,
                                 float* __restrict__ xa_f,
                                 _Float16* __restrict__ xa_h,
                                 int s) {
    int idx = blockIdx.x * blockDim.x + threadIdx.x;
    if (idx >= MROWS * ED) return;
    int e   = idx & (ED - 1);
    int row = idx >> 9;
    int b   = row / LL;
    int p   = row - b * LL;

    float acc = cb[e];
#pragma unroll
    for (int k = 0; k < KK; ++k) {
        int pp = p - (KK - 1) + k;
        if (pp >= 0) {
            int lo = perm_idx(s, pp);
            acc += cw[e * KK + k] *
                   xz[((size_t)(b * LL + lo)) * (2 * ED) + e];
        }
    }
    float o = acc / (1.0f + __expf(-acc));
    xa_f[idx] = o;
    xa_h[idx] = (_Float16)o;
}

// ---------------- selective scan, all 4 directions fused ----------------
struct ScanP {
    const float* Alog[4];
    const float* Dp[4];
};

__global__ void scan_kernel(const float* __restrict__ delta,   // [4][M][ED]
                            const float* __restrict__ xa,      // [4][M][ED]
                            const float* __restrict__ dBC,     // [4][M][48]
                            const float* __restrict__ xz,      // [M][2*ED]
                            ScanP P,
                            float* __restrict__ gbuf) {        // [4][M][ED]
    int g = blockIdx.x * blockDim.x + threadIdx.x;
    int ch = g >> 4;
    int n  = g & 15;
    if (ch >= 4 * B0 * ED) return;
    int s = ch >> 10;
    int b = (ch >> 9) & 1;
    int e = ch & (ED - 1);

    const float* del = delta + (size_t)s * MROWS * ED;
    const float* xav = xa    + (size_t)s * MROWS * ED;
    const float* bc  = dBC   + (size_t)s * MROWS * 48;
    float*       go  = gbuf  + (size_t)s * MROWS * ED;

    const float An = -__expf(P.Alog[s][e * NN + n]);
    const float De = P.Dp[s][e];
    float h = 0.0f;

    for (int l = 0; l < LL; ++l) {
        int row = b * LL + l;
        float dl = del[(size_t)row * ED + e];
        float xv = xav[(size_t)row * ED + e];
        float Bn = bc[(size_t)row * 48 + RR + n];
        float Cn = bc[(size_t)row * 48 + RR + NN + n];
        float dA = __expf(dl * An);
        h = dA * h + (dl * xv) * Bn;
        float t = h * Cn;
        t += __shfl_xor(t, 1, 16);
        t += __shfl_xor(t, 2, 16);
        t += __shfl_xor(t, 4, 16);
        t += __shfl_xor(t, 8, 16);
        if (n == 0) {
            float y  = t + De * xv;
            int lo   = perm_idx(s, l);        // involution: scatter target
            float za = xz[((size_t)(b * LL + lo)) * (2 * ED) + ED + e];
            go[((size_t)(b * LL + lo)) * ED + e] =
                y * (za / (1.0f + __expf(-za)));
        }
    }
}

// ---------------- host orchestration ----------------
extern "C" void kernel_launch(void* const* d_in, const int* in_sizes, int n_in,
                              void* d_out, int out_size, void* d_ws, size_t ws_size,
                              hipStream_t stream) {
    (void)in_sizes; (void)n_in; (void)out_size; (void)ws_size;

    const float* x     = (const float*)d_in[0];
    const float* in_w  = (const float*)d_in[1];
    const float* out_w = (const float*)d_in[2];

    char* wsc = (char*)d_ws;
    size_t off = 0;
    auto alloc = [&](size_t bytes) -> void* {
        void* p = wsc + off;
        off = (off + bytes + 255) & ~(size_t)255;
        return p;
    };

    const size_t ME  = (size_t)MROWS * ED;     // per-branch activation stride
    const size_t M48 = (size_t)MROWS * 48;

    _Float16* xh    = (_Float16*)alloc((size_t)MROWS * DM * 2);
    _Float16* inwh  = (_Float16*)alloc((size_t)(2 * ED) * DM * 2);
    _Float16* outwh = (_Float16*)alloc((size_t)DM * ED * 2);
    _Float16* xpwh  = (_Float16*)alloc((size_t)48 * ED * 2);    // reused/branch
    _Float16* dtwh  = (_Float16*)alloc((size_t)ED * 32 * 2);    // reused/branch
    float*    xz    = (float*)   alloc((size_t)MROWS * 2 * ED * 4);
    float*    xa_f  = (float*)   alloc(4 * ME * 4);             // per-branch
    _Float16* xa_h  = (_Float16*)alloc(ME * 2);                 // reused/branch
    float*    dBC   = (float*)   alloc(4 * M48 * 4);            // per-branch
    _Float16* dinh  = (_Float16*)alloc((size_t)MROWS * 32 * 2); // reused/branch
    float*    delta = (float*)   alloc(4 * ME * 4);             // per-branch
    float*    gbuf  = (float*)   alloc(4 * ME * 4);             // per-branch
    _Float16* yh    = (_Float16*)alloc(ME * 2);
    (void)alloc(65536);   // tail pad: TDM may touch past last M tile

    auto cdiv = [](int a, int b) { return (a + b - 1) / b; };
    const int MT = cdiv(MROWS, 16);            // 129 M-tiles

    // ---- f32 -> f16 staging ----
    int n;
    n = MROWS * DM;
    cvt_f32_f16<<<cdiv(n, 256), 256, 0, stream>>>(x, xh, n);
    n = 2 * ED * DM;
    cvt_f32_f16<<<cdiv(n, 256), 256, 0, stream>>>(in_w, inwh, n);
    n = DM * ED;
    cvt_f32_f16<<<cdiv(n, 256), 256, 0, stream>>>(out_w, outwh, n);

    // ---- in-projection: xz = x @ in_w.T  (2052 x 1024 x 256), TDM path ----
    gemm_wmma_tdm<<<dim3((2 * ED) / 256, MT), 128, 0, stream>>>(
        xh, inwh, xz, MROWS, 2 * ED, DM, nullptr, 0);

    // ---- per-branch projections (outputs to per-branch buffers) ----
    for (int s = 0; s < 4; ++s) {
        const float* conv_w = (const float*)d_in[3 + 7 * s + 0];
        const float* conv_b = (const float*)d_in[3 + 7 * s + 1];
        const float* xp_w   = (const float*)d_in[3 + 7 * s + 2];
        const float* dt_w   = (const float*)d_in[3 + 7 * s + 3];
        const float* dt_b   = (const float*)d_in[3 + 7 * s + 4];

        n = 48 * ED;
        cvt_f32_f16<<<cdiv(n, 256), 256, 0, stream>>>(xp_w, xpwh, n);
        n = ED * 32;
        pad_f32_f16<<<cdiv(n, 256), 256, 0, stream>>>(dt_w, dtwh, ED, RR, RR, 32);

        n = MROWS * ED;
        conv_silu_kernel<<<cdiv(n, 256), 256, 0, stream>>>(
            xz, conv_w, conv_b, xa_f + s * ME, xa_h, s);

        // dBC = xa @ xp_w.T   (2052 x 48 x 512): N=48 -> simple 16x16 path
        gemm_wmma_f16<<<dim3(48 / 16, MT), 32, 0, stream>>>(
            xa_h, xpwh, dBC + s * M48, MROWS, 48, ED, nullptr, 0);

        // delta = softplus(dBC[:, :16] @ dt_w.T + dt_b)  (K padded 16->32)
        n = MROWS * 32;
        pad_f32_f16<<<cdiv(n, 256), 256, 0, stream>>>(
            dBC + s * M48, dinh, MROWS, 48, RR, 32);
        gemm_wmma_tdm<<<dim3(ED / 256, MT), 128, 0, stream>>>(
            dinh, dtwh, delta + s * ME, MROWS, ED, 32, dt_b, 1);
    }

    // ---- fused 4-direction selective scan (2048 waves in flight) ----
    ScanP P;
    for (int s = 0; s < 4; ++s) {
        P.Alog[s] = (const float*)d_in[3 + 7 * s + 5];
        P.Dp[s]   = (const float*)d_in[3 + 7 * s + 6];
    }
    n = 4 * B0 * ED * NN;                      // 65536 threads
    scan_kernel<<<cdiv(n, 256), 256, 0, stream>>>(delta, xa_f, dBC, xz, P, gbuf);

    // ---- combine 4 directions, out-projection ----
    n = (int)ME;
    combine_cvt<<<cdiv(n, 256), 256, 0, stream>>>(gbuf, yh, n);
    gemm_wmma_tdm<<<dim3(DM / 256, MT), 128, 0, stream>>>(
        yh, outwh, (float*)d_out, MROWS, DM, ED, nullptr, 0);
}
